// Memory_Network_41961830482588
// MI455X (gfx1250) — matrix operand order repaired and checked by reading, not compiled, not run on hardware
//
#include <hip/hip_runtime.h>
#include <hip/hip_bf16.h>

typedef __attribute__((ext_vector_type(16))) _Float16 v16h;
typedef __attribute__((ext_vector_type(8)))  float    v8f;

#define VOCAB 50000
#define EMB   300
#define EPAD  320     // E padded to multiple of 32 for WMMA K
#define HID   512
#define G4H   2048    // 4*HID
#define IMG   4096
#define RR    10
#define LQ    20
#define LH    40
#define NR    1280    // B*R
#define CATK  4608    // IMG+HID

#define COLSTRIDE 20          // dwords per B-column in LDS (bank-conflict-free, 16B aligned)
#define BUFSTRIDE (64 * COLSTRIDE)   // dwords per LDS buffer (64 columns)

union Frag { v16h h; unsigned u[8]; };

#if __has_builtin(__builtin_amdgcn_global_load_async_to_lds_b32)
#define HAVE_ASYNC_LDS 1
typedef __attribute__((address_space(1))) int GI32;
typedef __attribute__((address_space(3))) int LI32;
#else
#define HAVE_ASYNC_LDS 0
#endif

__device__ __forceinline__ void wait_async_lds() {
#if HAVE_ASYNC_LDS
#if __has_builtin(__builtin_amdgcn_s_wait_asynccnt)
    __builtin_amdgcn_s_wait_asynccnt(0);
#else
    asm volatile("s_wait_asynccnt 0x0" ::: "memory");
#endif
#endif
}

// ---------------- utility: zero a u32 region ----------------
__global__ void k_zero(unsigned* __restrict__ p, long n) {
    long i = (long)blockIdx.x * blockDim.x + threadIdx.x;
    if (i < n) p[i] = 0u;
}

// ---------------- pack W[N,K] (f32) -> B[k,n]=W[n,k], 2 f16 per u32 at [k/2][n] ----------------
__global__ void k_pack_wT(const float* __restrict__ W, unsigned* __restrict__ out,
                          int N, int K, int Kpad) {
    long i = (long)blockIdx.x * blockDim.x + threadIdx.x;
    long tot = (long)N * (Kpad >> 1);
    if (i >= tot) return;
    int n  = (int)(i % N);
    int kp = (int)(i / N);
    int k0 = kp * 2, k1 = k0 + 1;
    union { _Float16 h[2]; unsigned u; } pk;
    pk.h[0] = (k0 < K) ? (_Float16)W[(long)n * K + k0] : (_Float16)0.f;
    pk.h[1] = (k1 < K) ? (_Float16)W[(long)n * K + k1] : (_Float16)0.f;
    out[(long)kp * N + n] = pk.u;
}

// ---------------- embedding gather, time-major [T][NR][EPAD], f16 ----------------
__global__ void k_embed(const int* __restrict__ toks, const float* __restrict__ emb,
                        _Float16* __restrict__ out, int T) {
    long i = (long)blockIdx.x * blockDim.x + threadIdx.x;
    long tot = (long)T * NR * EPAD;
    if (i >= tot) return;
    int e = (int)(i % EPAD);
    long nt = i / EPAD;
    int n = (int)(nt % NR);
    int t = (int)(nt / NR);
    int tok = toks[(long)n * T + t];
    float v = 0.f;
    if (e < EMB && tok != 0) v = emb[(long)tok * EMB + e];
    out[i] = (_Float16)v;
}

// ================= WMMA GEMM =================
// C[M,N] = A1[M,K1]*B1 + A2[M,K2]*B2
// A: row-major f16; B: packed u32 [K/2][N] (two f16 K-values per dword).
// Block = 256 threads = 8 waves; block tile 128(M) x 64(N); wave tile 16x64.
// B k-chunks staged in LDS (transposed, stride-20 pad) via async global->LDS,
// double-buffered; chunk loop unrolled x2 with alternating A-fragment registers
// so there is no acur=anext copy (avoids WMMA WAR hazard NOPs).
__global__ __launch_bounds__(256) void k_gemm(
        const _Float16* __restrict__ A1, int lda1, int K1, const unsigned* __restrict__ B1,
        const _Float16* __restrict__ A2, int lda2, int K2, const unsigned* __restrict__ B2,
        float* __restrict__ C, int M, int N) {
    __shared__ unsigned ldsB[2 * BUFSTRIDE];   // 10 KB

    const int lane = threadIdx.x & 31;
    const int wave = threadIdx.x >> 5;
    const int half = lane >> 4;
    const int ln   = lane & 15;
    const int m0   = (blockIdx.x * 8 + wave) << 4;
    const int n0   = blockIdx.y << 6;

    const int nch1 = K1 >> 5;
    const int nch  = nch1 + (K2 >> 5);

    // stage one 16x64-dword packed-B tile (chunk c) into LDS buffer `buf`
    auto stage = [&](int c, int buf) {
        const unsigned* Bp;
        int kb;
        if (c < nch1) { Bp = B1; kb = c << 4; }
        else          { Bp = B2; kb = (c - nch1) << 4; }
#pragma unroll
        for (int q = 0; q < 4; ++q) {
            int e   = threadIdx.x + (q << 8);     // 0..1023
            int col = e >> 4;                      // 0..63
            int kl  = e & 15;                      // 0..15
            const unsigned* g = Bp + (long)(kb + kl) * N + n0 + col;
            unsigned lidx = buf * BUFSTRIDE + col * COLSTRIDE + kl;
#if HAVE_ASYNC_LDS
            __builtin_amdgcn_global_load_async_to_lds_b32((GI32*)g, (LI32*)&ldsB[lidx], 0, 0);
#else
            ldsB[lidx] = *g;
#endif
        }
    };

    // per-lane A fragment load (16-bit A 16x32 layout) for chunk c
    auto loadA = [&](int c) -> Frag {
        const _Float16* A; int lda, kc;
        if (c < nch1) { A = A1; lda = lda1; kc = c << 5; }
        else          { A = A2; lda = lda2; kc = (c - nch1) << 5; }
        const unsigned* aRow = (const unsigned*)A + (((long)(m0 + ln) * lda) >> 1);
        const int kb = kc >> 1;
        Frag a;
#pragma unroll
        for (int v = 0; v < 8; ++v)
            a.u[v] = aRow[kb + (half << 3) + v];
        return a;
    };

    v8f acc[4] = {};

    // preload all 4 B fragments (one ds clause, one wait), then 4 back-to-back wmma
    auto compute = [&](int bufofs, const Frag& a) {
        Frag b[4];
#pragma unroll
        for (int j = 0; j < 4; ++j) {
            const uint4* p = (const uint4*)&ldsB[bufofs + half * 8 + (j * 16 + ln) * COLSTRIDE];
            uint4 x0 = p[0], x1 = p[1];            // two ds_load_b128, conflict-free
            b[j].u[0] = x0.x; b[j].u[1] = x0.y; b[j].u[2] = x0.z; b[j].u[3] = x0.w;
            b[j].u[4] = x1.x; b[j].u[5] = x1.y; b[j].u[6] = x1.z; b[j].u[7] = x1.w;
        }
#pragma unroll
        for (int j = 0; j < 4; ++j)
            acc[j] = __builtin_amdgcn_wmma_f32_16x16x32_f16(false, a.h, false, b[j].h,
                                                            (short)0, acc[j], false, false);
    };

    Frag a0 = loadA(0), a1;
    stage(0, 0);

    for (int c = 0; c < nch; c += 2) {
        wait_async_lds();
        __syncthreads();                           // chunk c staged in buf0
        if (c + 1 < nch) { stage(c + 1, 1); a1 = loadA(c + 1); }
        compute(0, a0);
        if (c + 1 >= nch) break;                   // uniform; nch is even in practice

        wait_async_lds();
        __syncthreads();                           // chunk c+1 staged in buf1
        if (c + 2 < nch) { stage(c + 2, 0); a0 = loadA(c + 2); }
        compute(BUFSTRIDE, a1);
    }

#pragma unroll
    for (int j = 0; j < 4; ++j) {
        int col = n0 + j * 16 + ln;
#pragma unroll
        for (int i = 0; i < 8; ++i)                // f32 C layout: M = i + 8*half, N = ln
            C[(long)(m0 + i + (half << 3)) * N + col] = acc[j][i];
    }
}

// ---------------- LSTM pointwise cell (PyTorch gate order i,f,g,o) ----------------
__device__ __forceinline__ float sigm(float x) { return 1.f / (1.f + __expf(-x)); }

__global__ void k_lstm_pw(const float* __restrict__ g, const float* __restrict__ bih,
                          const float* __restrict__ bhh,
                          float* __restrict__ c, float* __restrict__ h32,
                          _Float16* __restrict__ h16) {
    int idx = blockIdx.x * blockDim.x + threadIdx.x;
    if (idx >= NR * HID) return;
    int m = idx / HID, j = idx % HID;
    const float* gr = g + (long)m * G4H;
    float gi = gr[j]           + bih[j]           + bhh[j];
    float gf = gr[HID + j]     + bih[HID + j]     + bhh[HID + j];
    float gg = gr[2 * HID + j] + bih[2 * HID + j] + bhh[2 * HID + j];
    float go = gr[3 * HID + j] + bih[3 * HID + j] + bhh[3 * HID + j];
    float cn = sigm(gf) * c[idx] + sigm(gi) * tanhf(gg);
    c[idx] = cn;
    float hn = sigm(go) * tanhf(cn);
    h32[idx] = hn;
    h16[idx] = (_Float16)hn;
}

// ---------------- build [img | hq] concat, f16 [NR, CATK] ----------------
__global__ void k_cat(const float* __restrict__ img, const _Float16* __restrict__ hq16,
                      _Float16* __restrict__ out) {
    long i = (long)blockIdx.x * blockDim.x + threadIdx.x;
    if (i >= (long)NR * CATK) return;
    int n = (int)(i / CATK), k = (int)(i % CATK);
    out[i] = (k < IMG) ? (_Float16)img[(long)(n / RR) * IMG + k]
                       : hq16[(long)n * HID + (k - IMG)];
}

__global__ void k_tanh_bias(const float* __restrict__ pre, const float* __restrict__ bias,
                            float* __restrict__ out) {
    int idx = blockIdx.x * blockDim.x + threadIdx.x;
    if (idx >= NR * HID) return;
    out[idx] = tanhf(pre[idx] + bias[idx % HID]);
}

// ---------------- causal attention: one wave per (b,r) ----------------
__global__ void k_attn(const float* __restrict__ Q, const float* __restrict__ hf,
                       float* __restrict__ att, _Float16* __restrict__ att16) {
    int br = blockIdx.x;
    int b = br / RR, r = br % RR;
    int lane = threadIdx.x;
    const float* q = Q + (long)br * HID;
    float sc[RR];
    for (int s = 0; s <= r; ++s) {
        const float* f = hf + (long)(b * RR + s) * HID;
        float p = 0.f;
        for (int h = lane; h < HID; h += 32) p += q[h] * f[h];
        for (int off = 16; off > 0; off >>= 1) p += __shfl_xor(p, off, 32);
        sc[s] = p;
    }
    float mx = sc[0];
    for (int s = 1; s <= r; ++s) mx = fmaxf(mx, sc[s]);
    float sum = 0.f;
    for (int s = 0; s <= r; ++s) { float e = __expf(sc[s] - mx); sc[s] = e; sum += e; }
    float inv = 1.f / sum;
    for (int h = lane; h < HID; h += 32) {
        float a = 0.f;
        for (int s = 0; s <= r; ++s) a += sc[s] * hf[(long)(b * RR + s) * HID + h];
        a *= inv;
        att[(long)br * HID + h] = a;
        att16[(long)br * HID + h] = (_Float16)a;
    }
}

__global__ void k_final(const float* __restrict__ q, const float* __restrict__ pre,
                        const float* __restrict__ b2, float* __restrict__ out) {
    int idx = blockIdx.x * blockDim.x + threadIdx.x;
    if (idx >= NR * HID) return;
    out[idx] = q[idx] + tanhf(pre[idx] + b2[idx % HID]);
}

extern "C" void kernel_launch(void* const* d_in, const int* in_sizes, int n_in,
                              void* d_out, int out_size, void* d_ws, size_t ws_size,
                              hipStream_t stream) {
    (void)in_sizes; (void)n_in; (void)out_size; (void)ws_size;
    const float* img  = (const float*)d_in[0];
    const int*   qtok = (const int*)d_in[1];
    const int*   htok = (const int*)d_in[2];
    const float* emb  = (const float*)d_in[3];
    const float* Wihq = (const float*)d_in[4];
    const float* Whhq = (const float*)d_in[5];
    const float* bihq = (const float*)d_in[6];
    const float* bhhq = (const float*)d_in[7];
    const float* Wihf = (const float*)d_in[8];
    const float* Whhf = (const float*)d_in[9];
    const float* bihf = (const float*)d_in[10];
    const float* bhhf = (const float*)d_in[11];
    const float* W1   = (const float*)d_in[12];
    const float* b1   = (const float*)d_in[13];
    const float* W2   = (const float*)d_in[14];
    const float* b2   = (const float*)d_in[15];

    char* wsp = (char*)d_ws;
    auto alloc = [&](size_t bytes) -> void* {
        void* p = (void*)wsp;
        wsp += (bytes + 255) & ~(size_t)255;
        return p;
    };
    _Float16* qe    = (_Float16*)alloc((size_t)LQ * NR * EPAD * 2);
    _Float16* fe    = (_Float16*)alloc((size_t)LH * NR * EPAD * 2);
    unsigned* WihqP = (unsigned*)alloc((size_t)(EPAD / 2) * G4H * 4);
    unsigned* WhhqP = (unsigned*)alloc((size_t)(HID / 2) * G4H * 4);
    unsigned* WihfP = (unsigned*)alloc((size_t)(EPAD / 2) * G4H * 4);
    unsigned* WhhfP = (unsigned*)alloc((size_t)(HID / 2) * G4H * 4);
    unsigned* W1P   = (unsigned*)alloc((size_t)(CATK / 2) * HID * 4);
    unsigned* W2P   = (unsigned*)alloc((size_t)(HID / 2) * HID * 4);
    float*    gates = (float*)alloc((size_t)NR * G4H * 4);
    float*    cq    = (float*)alloc((size_t)NR * HID * 4);
    float*    hq32  = (float*)alloc((size_t)NR * HID * 4);
    _Float16* hq16  = (_Float16*)alloc((size_t)NR * HID * 2);
    float*    cf    = (float*)alloc((size_t)NR * HID * 4);
    float*    hf32  = (float*)alloc((size_t)NR * HID * 4);
    _Float16* hf16  = (_Float16*)alloc((size_t)NR * HID * 2);
    _Float16* Acat  = (_Float16*)alloc((size_t)NR * CATK * 2);
    float*    query = (float*)alloc((size_t)NR * HID * 4);
    float*    att   = (float*)alloc((size_t)NR * HID * 4);
    _Float16* att16 = (_Float16*)alloc((size_t)NR * HID * 2);

    const int TPB = 256;
    auto nb = [](long n, int t) { return (unsigned)((n + t - 1) / t); };

    // ---- pack all weights into WMMA-B layout ----
    k_pack_wT<<<nb((long)G4H * (EPAD / 2), TPB), TPB, 0, stream>>>(Wihq, WihqP, G4H, EMB, EPAD);
    k_pack_wT<<<nb((long)G4H * (HID / 2), TPB), TPB, 0, stream>>>(Whhq, WhhqP, G4H, HID, HID);
    k_pack_wT<<<nb((long)G4H * (EPAD / 2), TPB), TPB, 0, stream>>>(Wihf, WihfP, G4H, EMB, EPAD);
    k_pack_wT<<<nb((long)G4H * (HID / 2), TPB), TPB, 0, stream>>>(Whhf, WhhfP, G4H, HID, HID);
    k_pack_wT<<<nb((long)HID * (CATK / 2), TPB), TPB, 0, stream>>>(W1, W1P, HID, CATK, CATK);
    k_pack_wT<<<nb((long)HID * (HID / 2), TPB), TPB, 0, stream>>>(W2, W2P, HID, HID, HID);

    // ---- embeddings (time-major) ----
    k_embed<<<nb((long)LQ * NR * EPAD, TPB), TPB, 0, stream>>>(qtok, emb, qe, LQ);
    k_embed<<<nb((long)LH * NR * EPAD, TPB), TPB, 0, stream>>>(htok, emb, fe, LH);

    dim3 gGate(NR / 128, G4H / 64);
    dim3 gHid(NR / 128, HID / 64);

    // ---- question LSTM ----
    k_zero<<<nb(NR * HID / 2, TPB), TPB, 0, stream>>>((unsigned*)hq16, NR * HID / 2);
    k_zero<<<nb(NR * HID, TPB), TPB, 0, stream>>>((unsigned*)cq, NR * HID);
    for (int t = 0; t < LQ; ++t) {
        k_gemm<<<gGate, 256, 0, stream>>>(qe + (size_t)t * NR * EPAD, EPAD, EPAD, WihqP,
                                          hq16, HID, HID, WhhqP, gates, NR, G4H);
        k_lstm_pw<<<nb(NR * HID, TPB), TPB, 0, stream>>>(gates, bihq, bhhq, cq, hq32, hq16);
    }

    // ---- query = tanh([img|hq] @ W1^T + b1) ----
    k_cat<<<nb((long)NR * CATK, TPB), TPB, 0, stream>>>(img, hq16, Acat);
    k_gemm<<<gHid, 256, 0, stream>>>(Acat, CATK, CATK, W1P,
                                     nullptr, 0, 0, nullptr, gates, NR, HID);
    k_tanh_bias<<<nb(NR * HID, TPB), TPB, 0, stream>>>(gates, b1, query);

    // ---- fact LSTM ----
    k_zero<<<nb(NR * HID / 2, TPB), TPB, 0, stream>>>((unsigned*)hf16, NR * HID / 2);
    k_zero<<<nb(NR * HID, TPB), TPB, 0, stream>>>((unsigned*)cf, NR * HID);
    for (int t = 0; t < LH; ++t) {
        k_gemm<<<gGate, 256, 0, stream>>>(fe + (size_t)t * NR * EPAD, EPAD, EPAD, WihfP,
                                          hf16, HID, HID, WhhfP, gates, NR, G4H);
        k_lstm_pw<<<nb(NR * HID, TPB), TPB, 0, stream>>>(gates, bihf, bhhf, cf, hf32, hf16);
    }

    // ---- causal attention over fact history ----
    k_attn<<<NR, 32, 0, stream>>>(query, hf32, att, att16);

    // ---- out = query + tanh(att @ W2^T + b2) ----
    k_gemm<<<gHid, 256, 0, stream>>>(att16, HID, HID, W2P,
                                     nullptr, 0, 0, nullptr, gates, NR, HID);
    k_final<<<nb(NR * HID, TPB), TPB, 0, stream>>>(query, gates, b2, (float*)d_out);
}
